// SJModel_29583734735507
// MI455X (gfx1250) — matrix operand order repaired and checked by reading, not compiled, not run on hardware
//
#include <hip/hip_runtime.h>

typedef __bf16 bf16;
typedef __attribute__((ext_vector_type(16))) __bf16 v16bf;
typedef __attribute__((ext_vector_type(8)))  __bf16 v8bf;
typedef __attribute__((ext_vector_type(8)))  float  v8f;

#define T_STEPS 8
#define BATCH   16
#define CIN     64
#define COUT    64
#define HH      64
#define WW      64
#define XCOLS   66   // WW + 2-column halo

// Fused implicit-GEMM 3x3 conv (bf16 WMMA, f32 accum) + multi-step LIF.
// One workgroup = one (b, h) output row: 64 C_out x 64 pixels, all T steps.
__global__ __launch_bounds__(512)
void snn_conv_lif_kernel(const float* __restrict__ x,
                         const float* __restrict__ w,
                         float* __restrict__ out)
{
    __shared__ bf16 Wbf[COUT * 9 * CIN];   // [cout][tap][cin], cin contiguous
    __shared__ bf16 Xt[3 * XCOLS * CIN];   // [row][col][cin],  cin contiguous

    const int tid   = threadIdx.x;
    const int lane  = tid & 31;
    const int wave  = tid >> 5;            // 16 waves
    const int mtile = wave >> 2;           // 0..3 -> C_out block of 16
    const int ntile = wave & 3;            // 0..3 -> pixel block of 16

    const int b = blockIdx.x / HH;
    const int h = blockIdx.x % HH;

    // ---- stage weights fp32 -> bf16 into LDS, layout [cout][tap][cin] ----
    for (int i = tid; i < COUT * CIN * 9; i += 512) {
        int cout = i / (CIN * 9);
        int rem  = i % (CIN * 9);
        int cin  = rem / 9;
        int tap  = rem % 9;
        Wbf[(cout * 9 + tap) * CIN + cin] = (bf16)w[i];
    }
    // ---- zero the two halo columns once (never rewritten) ----
    if (tid < 3 * CIN) {
        int r = tid / CIN, c = tid % CIN;
        Xt[(r * XCOLS + 0) * CIN + c]           = (bf16)0.f;
        Xt[(r * XCOLS + (XCOLS - 1)) * CIN + c] = (bf16)0.f;
    }
    __syncthreads();

    // Membrane potential, persistent in accumulator registers across T.
    v8f v = {};

    // Per-lane fragment coordinates (bf16 16x16x32 WMMA layouts).
    const int mrow  = lane & 15;            // A: M row within tile
    const int koffA = (lane >> 4) * 16;     // A: lanes>=16 hold K 16..31
    const int ncol  = lane & 15;            // B/D: N column
    const int koffB = (lane >> 4) * 16;     // B: lanes>=16 hold K 16..31
    const int mbase = mtile * 16 + 8 * (lane >> 4);  // D: rows r + mbase

    for (int t = 0; t < T_STEPS; ++t) {
        // ---- stage x halo tile (rows h-1..h+1) fp32 -> bf16 into LDS ----
        if (t > 0) __syncthreads();   // previous timestep done reading Xt
        for (int i = tid; i < CIN * 3 * WW; i += 512) {
            int c  = i / (3 * WW);            // w varies fastest -> coalesced
            int rr = (i % (3 * WW)) / WW;
            int ww = i % WW;
            int hh = h + rr - 1;
            float val = 0.f;
            if (hh >= 0 && hh < HH)
                val = x[((((size_t)t * BATCH + b) * CIN + c) * HH + hh) * WW + ww];
            Xt[(rr * XCOLS + 1 + ww) * CIN + c] = (bf16)val;
        }
        __syncthreads();

        // Speculative prefetch of next timestep's center rows into L2.
        if (t + 1 < T_STEPS && lane == 0) {
            const float* p =
                &x[((((size_t)(t + 1) * BATCH + b) * CIN + wave * 4) * HH + h) * WW];
            __builtin_prefetch(p, 0, 1);
        }

        // ---- implicit GEMM: z = sum over 9 taps of W_tap @ x_shift ----
        v8f z = {};
        #pragma unroll
        for (int tap = 0; tap < 9; ++tap) {
            const int dr = tap / 3;
            const int dc = tap % 3;
            const bf16* wrow = &Wbf[((mtile * 16 + mrow) * 9 + tap) * CIN];
            const bf16* xcol = &Xt[(dr * XCOLS + (ntile * 16 + ncol + dc)) * CIN];
            #pragma unroll
            for (int k2 = 0; k2 < 2; ++k2) {     // K = CIN = 2 x 32
                const int kb = k2 * 32;
                // A fragment: two contiguous 8-elem (16B) chunks per lane
                v8bf a0 = *(const v8bf*)&wrow[kb + koffA];
                v8bf a1 = *(const v8bf*)&wrow[kb + koffA + 8];
                v16bf a;
                #pragma unroll
                for (int i = 0; i < 8; ++i) { a[i] = a0[i]; a[i + 8] = a1[i]; }
                // B fragment: 16 contiguous bf16 along cin (32B LDS load)
                v16bf bf = *(const v16bf*)&xcol[kb + koffB];
                z = __builtin_amdgcn_wmma_f32_16x16x32_bf16(
                        false, a, false, bf, (short)0, z, false, false);
            }
        }

        // ---- LIF: charge (tau=2), fire (Heaviside), hard reset ----
        float s[8];
        #pragma unroll
        for (int i = 0; i < 8; ++i) {
            float vi = v[i] + (z[i] - v[i]) * 0.5f;   // v += (z - v)/tau
            float si = (vi >= 1.0f) ? 1.0f : 0.0f;    // spike on v >= v_th
            v[i] = (si != 0.f) ? 0.f : vi;            // hard reset to 0
            s[i] = si;
        }

        // ---- store spikes: D layout (VGPR r, lane) -> (cout, pixel) ----
        float* o = &out[((((size_t)t * BATCH + b) * COUT + mbase) * HH + h) * WW
                        + (ntile * 16 + ncol)];
        #pragma unroll
        for (int r = 0; r < 8; ++r)
            o[(size_t)r * HH * WW] = s[r];
    }
}

extern "C" void kernel_launch(void* const* d_in, const int* in_sizes, int n_in,
                              void* d_out, int out_size, void* d_ws, size_t ws_size,
                              hipStream_t stream) {
    const float* x = (const float*)d_in[0];   // [T,B,Cin,H,W] fp32
    const float* w = (const float*)d_in[1];   // [Cout,Cin,3,3] fp32
    float* out = (float*)d_out;               // [T,B,Cout,H,W] fp32

    dim3 grid(BATCH * HH);   // 1024 workgroups: one per (b, h) row
    dim3 block(512);         // 16 wave32s: 4 C_out tiles x 4 pixel tiles
    hipLaunchKernelGGL(snn_conv_lif_kernel, grid, block, 0, stream, x, w, out);
}
